// GraphChannelMixer_55113020342547
// MI455X (gfx1250) — matrix operand order repaired and checked by reading, not compiled, not run on hardware
//
#include <hip/hip_runtime.h>

// ---------------------------------------------------------------------------
// GraphChannelMixer on gfx1250 (CDNA5): fused kernel, 2 graphs per workgroup.
// B=8, N=19, T=1000, D=512, 2 layers. bf16 WMMA (16x16x32) with fp32 accum.
// W^T K-tiles staged LDS-direct via global_load_async_to_lds_b128 (ASYNCcnt),
// double-buffered, and SHARED by both graphs (halves W L2 traffic).
// ---------------------------------------------------------------------------

typedef __attribute__((ext_vector_type(16))) __bf16 v16bf;
typedef __attribute__((ext_vector_type(8)))  float  v8f;

union BF16Frag {
  v16bf v;
  unsigned int u32[8];
  unsigned short u16[16];
};

__device__ __forceinline__ unsigned short f2bf(float f) {
  unsigned int u = __float_as_uint(f);
  u += 0x7FFFu + ((u >> 16) & 1u);   // round-to-nearest-even
  return (unsigned short)(u >> 16);
}

// LDS byte offsets (total 282624 B):
#define OFF_XF(gi)   ((gi) * 40960)            // 2 x 20x512 f32 (row 19 = dummy)
#define OFF_WT       81920                     // 2 x 512x32 bf16 (double buffer)
#define OFF_XT(gi)   (147456 + (gi) * 32768)   // 2 x 512x32 bf16 x^T
#define OFF_XMX(gi)  (212992 + (gi) * 32768)   // 2 x 32x512 bf16 x_mixed
#define OFF_ADJ(gi)  (278528 + (gi) * 2048)    // 2 x 32x32  bf16 adj_norm

// A-matrix 16x32 bf16 fragment (ISA 7.12.2): lane<16 -> K base 0, lane>=16 -> +8.
// VGPR i holds K pair base 2*(i&3) + 16*(i>>2). Source is M x K row-major.
__device__ __forceinline__ v16bf load_a_frag(const unsigned short* base, int M,
                                             int ldk, int k0, int halfSel) {
  BF16Frag f;
  const unsigned short* p = base + M * ldk + k0 + 8 * halfSel;
#pragma unroll
  for (int i = 0; i < 8; ++i) {
    int kb = 2 * (i & 3) + 16 * (i >> 2);
    f.u32[i] = *(const unsigned int*)(p + kb);
  }
  return f.v;
}

// B-matrix 32x16 bf16 fragment (same K order as A): lane>=16 -> +8,
// VGPR i holds K pair base 2*(i&3) + 16*(i>>2). Source stored transposed: N x K row-major.
__device__ __forceinline__ v16bf load_b_frag(const unsigned short* baseT, int N,
                                             int ldk, int k0, int halfSel) {
  BF16Frag f;
  const unsigned short* p = baseT + N * ldk + k0 + 8 * halfSel;
#pragma unroll
  for (int i = 0; i < 8; ++i) f.u32[i] = *(const unsigned int*)(p + 2 * (i & 3) + 16 * (i >> 2));
  return f.v;
}

// Async LDS-direct 16B load: GLOBAL_LOAD_ASYNC_TO_LDS_B128, GVS addressing.
__device__ __forceinline__ void async_lds_load_b128(unsigned lds_byte_addr,
                                                    unsigned gl_byte_off,
                                                    unsigned long long gbase) {
  asm volatile("global_load_async_to_lds_b128 %0, %1, %2"
               :
               : "v"(lds_byte_addr), "v"(gl_byte_off), "s"(gbase)
               : "memory");
}

// ---------------------------------------------------------------------------
// Prep: convert W (2,512,512) fp32 -> bf16 W^T, layout [layer][n][k].
// ---------------------------------------------------------------------------
__global__ void prep_w_kernel(const float* __restrict__ W,
                              unsigned short* __restrict__ wsW) {
  int idx = blockIdx.x * 256 + threadIdx.x;   // idx = ((l*512 + n)*512 + k)
  int l = idx >> 18;
  int rem = idx & 0x3FFFF;
  int n = rem >> 9;
  int k = rem & 511;
  wsW[idx] = f2bf(W[((size_t)l * 512 + k) * 512 + n]);
}

// ---------------------------------------------------------------------------
// Main fused kernel: one workgroup (512 thr = 16 waves) per PAIR of graphs.
// ---------------------------------------------------------------------------
__global__ __launch_bounds__(512, 1)
void mixer_kernel(const float* __restrict__ features,
                  const float* __restrict__ adjacency,
                  const float* __restrict__ edge_w,
                  const float* __restrict__ edge_b,
                  const unsigned short* __restrict__ wsW,   // bf16 W^T [2][512][512]
                  const float* __restrict__ bias,
                  const float* __restrict__ gamma,
                  const float* __restrict__ beta,
                  float* __restrict__ out) {
  extern __shared__ char smem[];

  const int tid     = threadIdx.x;
  const int lane    = tid & 31;
  const int wave    = tid >> 5;           // 16 waves
  const int halfSel = lane >> 4;
  const int col     = lane & 15;
  const int g2 = blockIdx.x;              // graph pair; graphs 2*g2, 2*g2+1

  const unsigned wt_lds_base = (unsigned)(size_t)(smem + OFF_WT);  // AS3 offset
  const unsigned long long wbase = (unsigned long long)(size_t)wsW;

  // ---- zero the K-pad columns (19..31) of x^T once; never rewritten
#pragma unroll
  for (int gi = 0; gi < 2; ++gi) {
    unsigned short* xT = (unsigned short*)(smem + OFF_XT(gi));
    for (int iq = tid; iq < 512 * 13; iq += 512) {
      int d = iq / 13;
      int k = 19 + (iq - d * 13);
      xT[d * 32 + k] = 0;
    }
  }

  // ---- load activations: (b, n, t, :) -> xf32 row-major + bf16 x^T
#pragma unroll
  for (int gi = 0; gi < 2; ++gi) {
    float*          xf = (float*)(smem + OFF_XF(gi));
    unsigned short* xT = (unsigned short*)(smem + OFF_XT(gi));
    int g  = 2 * g2 + gi;                 // b*1000 + t  (1000 even: same pair b)
    int bI = g / 1000;
    int tI = g - bI * 1000;
    for (int idx = tid; idx < 19 * 128; idx += 512) {
      int n  = idx >> 7;
      int dc = (idx & 127) << 2;
      float4 f4 = *(const float4*)(features +
                   (((size_t)(bI * 19 + n) * 1000 + tI) * 512 + dc));
      *(float4*)(xf + n * 512 + dc) = f4;
      xT[(dc + 0) * 32 + n] = f2bf(f4.x);
      xT[(dc + 1) * 32 + n] = f2bf(f4.y);
      xT[(dc + 2) * 32 + n] = f2bf(f4.z);
      xT[(dc + 3) * 32 + n] = f2bf(f4.w);
    }
  }

  // ---- adjacency: softplus(a*w+b), row-normalize -> 32x32 bf16 (zero-padded)
  if (tid < 38) {
    int gi = (tid >= 19) ? 1 : 0;
    int r  = tid - gi * 19;
    int gg = 2 * g2 + gi;
    unsigned short* adjp = (unsigned short*)(smem + OFF_ADJ(gi));
    float ew = edge_w[0], eb = edge_b[0];
    const float* arow = adjacency + ((size_t)gg * 19 + r) * 19;
    float aw[19];
    float s = 0.f;
#pragma unroll
    for (int j = 0; j < 19; ++j) {
      float z  = arow[j] * ew + eb;
      float sp = (z > 20.f) ? z : log1pf(expf(z));
      aw[j] = sp;
      s += sp;
    }
    float inv = 1.f / (s + 1e-6f);
#pragma unroll
    for (int j = 0; j < 32; ++j)
      adjp[r * 32 + j] = (j < 19) ? f2bf(aw[j] * inv) : (unsigned short)0;
  }
  __syncthreads();

  const int mt = wave >> 3;   // M-tile (rows 0..15 / 16..31)
  const int q  = wave & 7;    // 8 waves per M-tile cover 32 N-tiles

  for (int layer = 0; layer < 2; ++layer) {
    // ---- kick off async DMA of W^T tile 0 into buf0; overlaps with GEMM1
#pragma unroll
    for (int i = 0; i < 4; ++i) {
      int c = tid + 512 * i;          // 0..2047 : 512 rows x 4 x 16B
      int n = c >> 2, ch = c & 3;
      unsigned goff  = (((unsigned)(layer * 512 + n) * 512) << 1) + ch * 16;
      unsigned laddr = wt_lds_base + (unsigned)(n * 64 + ch * 16);
      async_lds_load_b128(laddr, goff, wbase);
    }

    // ===== GEMM1: x_nb = adj_norm @ x ; fused alpha-mix -> xmx (bf16 MxK)
#pragma unroll
    for (int gi = 0; gi < 2; ++gi) {
      const float*          xf  = (const float*)(smem + OFF_XF(gi));
      const unsigned short* xT  = (const unsigned short*)(smem + OFF_XT(gi));
      unsigned short*       xmx = (unsigned short*)(smem + OFF_XMX(gi));
      v16bf a = load_a_frag((const unsigned short*)(smem + OFF_ADJ(gi)),
                            16 * mt + col, 32, 0, halfSel);
#pragma unroll
      for (int t4 = 0; t4 < 4; ++t4) {
        int nt = q + 8 * t4;
        v16bf bb = load_b_frag(xT, nt * 16 + col, 32, 0, halfSel);
        v8f c = {};
        c = __builtin_amdgcn_wmma_f32_16x16x32_bf16(false, a, false, bb,
                                                    (short)0, c, false, false);
        int Nc = nt * 16 + col;
#pragma unroll
        for (int r = 0; r < 8; ++r) {
          int M  = 16 * mt + r + 8 * halfSel;
          int Mc = (M < 19) ? M : 19;          // branchless clamp to dummy row
          float xm = 0.95f * xf[Mc * 512 + Nc] + 0.05f * c[r];
          xmx[M * 512 + Nc] = f2bf(xm);
        }
      }
    }

    // ===== GEMM2: y = x_mixed @ W[layer], K=512, 16 double-buffered K-tiles;
    //       staged W tile shared by both graphs.
    v8f acc[2][4];
#pragma unroll
    for (int gi = 0; gi < 2; ++gi)
#pragma unroll
      for (int j = 0; j < 4; ++j) { v8f z = {}; acc[gi][j] = z; }

    for (int ks = 0; ks < 16; ++ks) {
      __syncthreads();   // everyone done reading buf[(ks+1)&1] from iter ks-1
      if (ks < 15) {
        int nk = ks + 1;
        unsigned lbuf = wt_lds_base + ((nk & 1) ? 32768u : 0u);
#pragma unroll
        for (int i = 0; i < 4; ++i) {
          int c = tid + 512 * i;
          int n = c >> 2, ch = c & 3;
          unsigned goff = (((unsigned)(layer * 512 + n) * 512 + nk * 32) << 1)
                          + ch * 16;
          async_lds_load_b128(lbuf + (unsigned)(n * 64 + ch * 16), goff, wbase);
        }
        asm volatile("s_wait_asynccnt 0x4" ::: "memory");  // tile ks complete
      } else {
        asm volatile("s_wait_asynccnt 0x0" ::: "memory");
      }
      __syncthreads();   // tile ks visible to all waves (also orders xmx @ks=0)

      const unsigned short* wt = (const unsigned short*)(smem + OFF_WT) +
                                 ((ks & 1) ? 16384 : 0);
      v16bf a0 = load_a_frag((const unsigned short*)(smem + OFF_XMX(0)),
                             16 * mt + col, 512, ks * 32, halfSel);
      v16bf a1 = load_a_frag((const unsigned short*)(smem + OFF_XMX(1)),
                             16 * mt + col, 512, ks * 32, halfSel);
#pragma unroll
      for (int j = 0; j < 4; ++j) {
        int nt = q * 4 + j;
        v16bf bb = load_b_frag(wt, nt * 16 + col, 32, 0, halfSel);
        acc[0][j] = __builtin_amdgcn_wmma_f32_16x16x32_bf16(false, a0, false, bb,
                                                            (short)0, acc[0][j],
                                                            false, false);
        acc[1][j] = __builtin_amdgcn_wmma_f32_16x16x32_bf16(false, a1, false, bb,
                                                            (short)0, acc[1][j],
                                                            false, false);
      }
    }

    // ===== epilogue: y + bias + residual -> xf32 (branchless row clamp)
    __syncthreads();
#pragma unroll
    for (int gi = 0; gi < 2; ++gi) {
      float* xf = (float*)(smem + OFF_XF(gi));
#pragma unroll
      for (int j = 0; j < 4; ++j) {
        int Nc = (q * 4 + j) * 16 + col;
        float bv = bias[layer * 512 + Nc];
#pragma unroll
        for (int r = 0; r < 8; ++r) {
          int M  = 16 * mt + r + 8 * halfSel;
          int Mc = (M < 19) ? M : 19;
          xf[Mc * 512 + Nc] += acc[gi][j][r] + bv;
        }
      }
    }
    __syncthreads();

    // ===== LayerNorm (population var) + exact-erf GELU; refresh bf16 x^T
#pragma unroll
    for (int gi = 0; gi < 2; ++gi) {
      float*          xf = (float*)(smem + OFF_XF(gi));
      unsigned short* xT = (unsigned short*)(smem + OFF_XT(gi));
      for (int row = wave; row < 19; row += 16) {
        float vals[16];
        float s = 0.f, sq = 0.f;
#pragma unroll
        for (int k = 0; k < 16; ++k) {
          float v = xf[row * 512 + k * 32 + lane];
          vals[k] = v;
          s += v;
          sq += v * v;
        }
#pragma unroll
        for (int off = 16; off > 0; off >>= 1) {
          s  += __shfl_xor(s, off, 32);
          sq += __shfl_xor(sq, off, 32);
        }
        float mu   = s * (1.f / 512.f);
        float var  = sq * (1.f / 512.f) - mu * mu;
        float rstd = rsqrtf(var + 1e-5f);
#pragma unroll
        for (int k = 0; k < 16; ++k) {
          int d = k * 32 + lane;
          float xn = (vals[k] - mu) * rstd * gamma[layer * 512 + d] +
                     beta[layer * 512 + d];
          float ge = 0.5f * xn * (1.f + erff(xn * 0.70710678118654752f));
          xf[row * 512 + d] = ge;
          xT[d * 32 + row]  = f2bf(ge);
        }
      }
    }
    __syncthreads();
  }

  // ---- writeback (B, N, T, D) layout, fp32
#pragma unroll
  for (int gi = 0; gi < 2; ++gi) {
    const float* xf = (const float*)(smem + OFF_XF(gi));
    int g  = 2 * g2 + gi;
    int bI = g / 1000;
    int tI = g - bI * 1000;
    for (int idx = tid; idx < 19 * 128; idx += 512) {
      int n  = idx >> 7;
      int dc = (idx & 127) << 2;
      *(float4*)(out + (((size_t)(bI * 19 + n) * 1000 + tI) * 512 + dc)) =
          *(const float4*)(xf + n * 512 + dc);
    }
  }
}

// ---------------------------------------------------------------------------
extern "C" void kernel_launch(void* const* d_in, const int* in_sizes, int n_in,
                              void* d_out, int out_size, void* d_ws,
                              size_t ws_size, hipStream_t stream) {
  const float* features  = (const float*)d_in[0];
  const float* adjacency = (const float*)d_in[1];
  const float* edge_w    = (const float*)d_in[2];
  const float* edge_b    = (const float*)d_in[3];
  const float* W         = (const float*)d_in[4];
  const float* bias      = (const float*)d_in[5];
  const float* gamma     = (const float*)d_in[6];
  const float* beta      = (const float*)d_in[7];

  unsigned short* wsW = (unsigned short*)d_ws;   // 2*512*512 bf16 = 1 MB

  prep_w_kernel<<<2048, 256, 0, stream>>>(W, wsW);

  const size_t lds_bytes = 282624;  // see LDS map above
  (void)hipFuncSetAttribute((const void*)mixer_kernel,
                            hipFuncAttributeMaxDynamicSharedMemorySize,
                            (int)lds_bytes);
  mixer_kernel<<<4000, 512, lds_bytes, stream>>>(
      features, adjacency, edge_w, edge_b, wsW, bias, gamma, beta,
      (float*)d_out);
}
